// MultiheadAttention_16192026706657
// MI455X (gfx1250) — matrix operand order repaired and checked by reading, not compiled, not run on hardware
//
#include <hip/hip_runtime.h>
#include <cstdint>

// ---------------------------------------------------------------------------
// MI455X (gfx1250) MHA forward. All GEMMs via v_wmma_f32_16x16x32_bf16
// (wave32, 16x16 tiles, f32 accumulate). bf16 conversion happens in LDS
// staging; WMMA fragment layouts follow CDNA5 ISA section 7.12.2.
// ---------------------------------------------------------------------------

typedef __attribute__((ext_vector_type(16))) __bf16 v16bf;
typedef __attribute__((ext_vector_type(8)))  float  v8f;

#define TILE_M 128
#define TILE_N 128
#define TILE_K 32

static constexpr int Bsz = 4, Lsz = 2048, Dsz = 2048;

__device__ __forceinline__ unsigned short f2bf(float f) {
    union { float f; unsigned u; } c; c.f = f;
    unsigned u = c.u;
    u += 0x7FFFu + ((u >> 16) & 1u);            // round-to-nearest-even
    return (unsigned short)(u >> 16);
}
__device__ __forceinline__ float bf2f(unsigned short h) {
    union { unsigned u; float f; } c; c.u = ((unsigned)h) << 16;
    return c.f;
}

union Frag32B { uint4 u[2]; v16bf v; };         // 16 bf16 = one WMMA A/B fragment

// ---------------------------------------------------------------------------
// Tiled GEMM: D = scale*(A @ B) [+ bias].  A: MxK, B: KxN, out: MxN.
//   AF32   : A source is f32 (converted to bf16 while staging), else bf16.
//   BF32   : B source is f32 laid out [k][n] (weights), converted+transposed.
//   B_NK   : (bf16 only) B source laid out [n][k] (i.e. K^T from row-major K).
//   OUT_BF16 / HAS_BIAS: epilogue options.
// Block tile 128x128x32; 8 waves; each wave = 2x4 wmma accumulators (32x64).
// ---------------------------------------------------------------------------
template<bool AF32, bool BF32, bool B_NK, bool OUT_BF16, bool HAS_BIAS>
__global__ __launch_bounds__(256) void gemm_wmma_bf16(
    const void* __restrict__ Aany, const void* __restrict__ Bany,
    const float* __restrict__ bias, void* __restrict__ Oany,
    int lda, int ldb, int ldo,
    long aBatch, long bBatch, long oBatch,
    int K, float scale)
{
    __shared__ unsigned short ldsA[TILE_M * TILE_K];   // [m][k] row-major, bf16
    __shared__ unsigned short ldsB[TILE_N * TILE_K];   // [n][k] (K contiguous), bf16

    const int tid  = threadIdx.x;
    const int lane = tid & 31;
    const int wave = tid >> 5;
    const int wm   = wave & 3;          // 4 waves along M: 4*32 = 128 rows
    const int wn   = wave >> 2;         // 2 waves along N: 2*64 = 128 cols
    const int half = lane >> 4;         // lane group (ISA fragment layout)
    const int l15  = lane & 15;

    const int m0 = blockIdx.y * TILE_M;
    const int n0 = blockIdx.x * TILE_N;
    const int bz = blockIdx.z;

    const float*          Af = (const float*)Aany          + (size_t)bz * aBatch;
    const unsigned short* Ah = (const unsigned short*)Aany + (size_t)bz * aBatch;
    const float*          Bf = (const float*)Bany          + (size_t)bz * bBatch;
    const unsigned short* Bh = (const unsigned short*)Bany + (size_t)bz * bBatch;

    v8f acc[2][4];
    const v8f vzero = {0.f, 0.f, 0.f, 0.f, 0.f, 0.f, 0.f, 0.f};
    #pragma unroll
    for (int mi = 0; mi < 2; ++mi)
        #pragma unroll
        for (int ni = 0; ni < 4; ++ni) acc[mi][ni] = vzero;

    for (int kk = 0; kk < K; kk += TILE_K) {
        // ---------------- stage A tile (128x32) into LDS as bf16 ------------
        if constexpr (AF32) {
            if (kk + TILE_K < K)   // global_prefetch_b8 hint for next K tile
                __builtin_prefetch(Af + (size_t)(m0 + (tid >> 1)) * lda + kk + TILE_K, 0, 1);
            #pragma unroll
            for (int i = 0; i < 4; ++i) {
                int q   = tid + i * 256;       // 1024 float4 chunks
                int row = q >> 3;
                int kc  = (q & 7) * 4;
                const float4 v = *(const float4*)(Af + (size_t)(m0 + row) * lda + kk + kc);
                ushort4 h;
                h.x = f2bf(v.x); h.y = f2bf(v.y); h.z = f2bf(v.z); h.w = f2bf(v.w);
                *(ushort4*)(ldsA + row * TILE_K + kc) = h;
            }
        } else {
            if (kk + TILE_K < K)
                __builtin_prefetch(Ah + (size_t)(m0 + (tid >> 1)) * lda + kk + TILE_K, 0, 1);
            #pragma unroll
            for (int i = 0; i < 2; ++i) {
                int q   = tid + i * 256;       // 512 uint4 chunks (8 bf16 each)
                int row = q >> 2;
                int c   = q & 3;
                ((uint4*)(ldsA + row * TILE_K))[c] =
                    ((const uint4*)(Ah + (size_t)(m0 + row) * lda + kk))[c];
            }
        }
        // ---------------- stage B tile into LDS as [n][k] bf16 --------------
        if constexpr (!BF32 && B_NK) {
            // Source already [n][k] (rows of K for Q@K^T): straight copy.
            #pragma unroll
            for (int i = 0; i < 2; ++i) {
                int q = tid + i * 256;
                int n = q >> 2;
                int c = q & 3;
                ((uint4*)(ldsB + n * TILE_K))[c] =
                    ((const uint4*)(Bh + (size_t)(n0 + n) * ldb + kk))[c];
            }
        } else if constexpr (!BF32 && !B_NK) {
            // Source [k][n] bf16 (V matrix): transpose into [n][k].
            #pragma unroll
            for (int i = 0; i < 2; ++i) {
                int q  = tid + i * 256;        // 512 chunks of 8 bf16
                int k  = q >> 4;
                int nc = (q & 15) * 8;
                uint4 u = *(const uint4*)(Bh + (size_t)(kk + k) * ldb + n0 + nc);
                unsigned short s[8];
                s[0] = u.x & 0xFFFFu; s[1] = u.x >> 16;
                s[2] = u.y & 0xFFFFu; s[3] = u.y >> 16;
                s[4] = u.z & 0xFFFFu; s[5] = u.z >> 16;
                s[6] = u.w & 0xFFFFu; s[7] = u.w >> 16;
                #pragma unroll
                for (int j = 0; j < 8; ++j) ldsB[(nc + j) * TILE_K + k] = s[j];
            }
        } else {
            // Source [k][n] f32 (weight matrices): convert + transpose.
            #pragma unroll
            for (int i = 0; i < 4; ++i) {
                int q  = tid + i * 256;        // 1024 float4 chunks
                int k  = q >> 5;
                int nc = (q & 31) * 4;
                const float4 f = *(const float4*)(Bf + (size_t)(kk + k) * ldb + n0 + nc);
                ldsB[(nc + 0) * TILE_K + k] = f2bf(f.x);
                ldsB[(nc + 1) * TILE_K + k] = f2bf(f.y);
                ldsB[(nc + 2) * TILE_K + k] = f2bf(f.z);
                ldsB[(nc + 3) * TILE_K + k] = f2bf(f.w);
            }
        }
        __syncthreads();

        // ---------------- fragment loads (ISA 7.12.2 layouts) + WMMA --------
        Frag32B a[2], b[4];
        #pragma unroll
        for (int mi = 0; mi < 2; ++mi) {
            // A 16x32: lane<16 holds row m=l15, K {0..7,16..23}; lane>=16 K {8..15,24..31}
            const unsigned short* p =
                ldsA + (wm * 32 + mi * 16 + l15) * TILE_K + half * 8;
            a[mi].u[0] = *(const uint4*)(p);
            a[mi].u[1] = *(const uint4*)(p + 16);
        }
        #pragma unroll
        for (int ni = 0; ni < 4; ++ni) {
            // B 32x16: lane<16 holds col n=l15, K {0..7,16..23}; lane>=16 K {8..15,24..31}
            const unsigned short* p =
                ldsB + (wn * 64 + ni * 16 + l15) * TILE_K + half * 8;
            b[ni].u[0] = *(const uint4*)(p);
            b[ni].u[1] = *(const uint4*)(p + 16);
        }
        #pragma unroll
        for (int mi = 0; mi < 2; ++mi)
            #pragma unroll
            for (int ni = 0; ni < 4; ++ni)
                acc[mi][ni] = __builtin_amdgcn_wmma_f32_16x16x32_bf16(
                    false, a[mi].v, false, b[ni].v,
                    (short)0, acc[mi][ni], false, false);
        __syncthreads();
    }

    // ---------------- epilogue: C/D layout VGPR r -> M=r+8*half, N=l15 ------
    float*          Of = (float*)Oany          + (size_t)bz * oBatch;
    unsigned short* Oh = (unsigned short*)Oany + (size_t)bz * oBatch;
    #pragma unroll
    for (int mi = 0; mi < 2; ++mi) {
        #pragma unroll
        for (int ni = 0; ni < 4; ++ni) {
            const int col = n0 + wn * 64 + ni * 16 + l15;
            const float bc = HAS_BIAS ? bias[col] : 0.0f;
            #pragma unroll
            for (int r = 0; r < 8; ++r) {
                const int row = m0 + wm * 32 + mi * 16 + half * 8 + r;
                const float v = acc[mi][ni][r] * scale + bc;
                if constexpr (OUT_BF16) Oh[(size_t)row * ldo + col] = f2bf(v);
                else                    Of[(size_t)row * ldo + col] = v;
            }
        }
    }
}

// ---------------------------------------------------------------------------
// Block-wide reductions (256 threads)
// ---------------------------------------------------------------------------
__device__ __forceinline__ float block_sum(float v, float* red) {
    const int tid = threadIdx.x;
    __syncthreads();
    red[tid] = v; __syncthreads();
    #pragma unroll
    for (int s = 128; s > 0; s >>= 1) {
        if (tid < s) red[tid] += red[tid + s];
        __syncthreads();
    }
    return red[0];
}
__device__ __forceinline__ float block_max(float v, float* red) {
    const int tid = threadIdx.x;
    __syncthreads();
    red[tid] = v; __syncthreads();
    #pragma unroll
    for (int s = 128; s > 0; s >>= 1) {
        if (tid < s) red[tid] = fmaxf(red[tid], red[tid + s]);
        __syncthreads();
    }
    return red[0];
}

// ---------------------------------------------------------------------------
// Row softmax, in place over attn logits (one block per row of 2048).
// ---------------------------------------------------------------------------
__global__ __launch_bounds__(256) void softmax_rows(float* __restrict__ attn) {
    __shared__ float red[256];
    const int tid = threadIdx.x;
    float* row = attn + (size_t)blockIdx.x * Lsz;
    float v[8];
    float m = -3.4e38f;
    #pragma unroll
    for (int i = 0; i < 8; ++i) { v[i] = row[tid + i * 256]; m = fmaxf(m, v[i]); }
    m = block_max(m, red);
    float s = 0.f;
    #pragma unroll
    for (int i = 0; i < 8; ++i) { v[i] = __expf(v[i] - m); s += v[i]; }
    s = block_sum(s, red);
    const float inv = 1.0f / s;
    #pragma unroll
    for (int i = 0; i < 8; ++i) row[tid + i * 256] = v[i] * inv;
}

// ---------------------------------------------------------------------------
// resid = out + Q[b, row 0, :] (reference quirk), then LayerNorm. In place.
// ---------------------------------------------------------------------------
__global__ __launch_bounds__(256) void resid_layernorm(
    float* __restrict__ y, const unsigned short* __restrict__ Qbf,
    const float* __restrict__ gamma, const float* __restrict__ beta)
{
    __shared__ float red[256];
    const int tid = threadIdx.x;
    const int row = blockIdx.x;            // 0..B*L-1
    const int bb  = row / Lsz;
    float* yr = y + (size_t)row * Dsz;
    const unsigned short* q0 = Qbf + (size_t)bb * Lsz * Dsz;  // row 0 of batch

    float v[8];
    float s = 0.f;
    #pragma unroll
    for (int i = 0; i < 8; ++i) {
        int d = tid + i * 256;
        v[i] = yr[d] + bf2f(q0[d]);
        s += v[i];
    }
    const float mu = block_sum(s, red) * (1.0f / Dsz);
    float vs = 0.f;
    #pragma unroll
    for (int i = 0; i < 8; ++i) { float t = v[i] - mu; vs += t * t; }
    const float var = block_sum(vs, red) * (1.0f / Dsz);
    const float rs  = rsqrtf(var + 1e-6f);
    #pragma unroll
    for (int i = 0; i < 8; ++i) {
        int d = tid + i * 256;
        yr[d] = (v[i] - mu) * rs * gamma[d] + beta[d];
    }
}

// ---------------------------------------------------------------------------
extern "C" void kernel_launch(void* const* d_in, const int* in_sizes, int n_in,
                              void* d_out, int out_size, void* d_ws, size_t ws_size,
                              hipStream_t stream) {
    const float* q_in = (const float*)d_in[0];
    const float* k_in = (const float*)d_in[1];
    const float* v_in = (const float*)d_in[2];
    const float* wq_w = (const float*)d_in[3];
    const float* wq_b = (const float*)d_in[4];
    const float* wk_w = (const float*)d_in[5];
    const float* wk_b = (const float*)d_in[6];
    const float* wv_w = (const float*)d_in[7];
    const float* wv_b = (const float*)d_in[8];
    const float* wo_w = (const float*)d_in[9];
    const float* wo_b = (const float*)d_in[10];
    const float* ln_g = (const float*)d_in[11];
    const float* ln_b = (const float*)d_in[12];

    const size_t BL  = (size_t)Bsz * Lsz;        // 8192 rows total
    const size_t BLD = BL * Dsz;                 // 16,777,216 elements

    // Workspace (bf16): Q, K, V, CTX -> 4 * 32 MiB = 128 MiB
    unsigned short* Qbf  = (unsigned short*)d_ws;
    unsigned short* Kbf  = Qbf + BLD;
    unsigned short* Vbf  = Kbf + BLD;
    unsigned short* CTXb = Vbf + BLD;

    float* y    = (float*)d_out;                 // [B,L,D] f32
    float* attn = y + BLD;                       // [B,1,L,L] f32

    const dim3 blk(256);
    const dim3 gProj (Dsz / TILE_N, (unsigned)(BL / TILE_M), 1);   // 16 x 64
    const dim3 gBatch(Dsz / TILE_N, Lsz / TILE_M, Bsz);            // 16 x 16 x 4

    const long  LD = (long)Lsz * Dsz;            // per-batch stride, elements
    const long  LL = (long)Lsz * Lsz;
    const float inv_sqrt_d = 0.022097086912079608f;   // 1/sqrt(2048)

    // 1) Q/K/V projections: X(f32) @ W(f32,[k][n]) + b -> bf16 workspace
    gemm_wmma_bf16<true,  true,  false, true,  true ><<<gProj, blk, 0, stream>>>(
        q_in, wq_w, wq_b, Qbf, Dsz, Dsz, Dsz, 0, 0, 0, Dsz, 1.0f);
    gemm_wmma_bf16<true,  true,  false, true,  true ><<<gProj, blk, 0, stream>>>(
        k_in, wk_w, wk_b, Kbf, Dsz, Dsz, Dsz, 0, 0, 0, Dsz, 1.0f);
    gemm_wmma_bf16<true,  true,  false, true,  true ><<<gProj, blk, 0, stream>>>(
        v_in, wv_w, wv_b, Vbf, Dsz, Dsz, Dsz, 0, 0, 0, Dsz, 1.0f);

    // 2) logits = scale * Q @ K^T per batch (B source is [n][k] = rows of K)
    gemm_wmma_bf16<false, false, true,  false, false><<<gBatch, blk, 0, stream>>>(
        Qbf, Kbf, nullptr, attn, Dsz, Dsz, Lsz, LD, LD, LL, Dsz, inv_sqrt_d);

    // 3) softmax over keys, in place (also the second reference output)
    softmax_rows<<<dim3((unsigned)BL), blk, 0, stream>>>(attn);

    // 4) ctx = attn(f32) @ V(bf16,[k][n]) -> bf16
    gemm_wmma_bf16<true,  false, false, true,  false><<<gBatch, blk, 0, stream>>>(
        attn, Vbf, nullptr, CTXb, Lsz, Dsz, Dsz, LL, LD, LD, Lsz, 1.0f);

    // 5) out = ctx @ wo + wo_b -> f32 directly into y region of d_out
    gemm_wmma_bf16<false, true,  false, false, true ><<<gProj, blk, 0, stream>>>(
        CTXb, wo_w, wo_b, y, Dsz, Dsz, Dsz, 0, 0, 0, Dsz, 1.0f);

    // 6) residual (+Q row0 per batch) + LayerNorm, in place on y
    resid_layernorm<<<dim3((unsigned)BL), blk, 0, stream>>>(y, Qbf, ln_g, ln_b);
}
